// SparseAutoencoder_35631048687836
// MI455X (gfx1250) — matrix operand order repaired and checked by reading, not compiled, not run on hardware
//
#include <hip/hip_runtime.h>

#define N_ROWS 16384
#define DIN    512
#define DDICT  8192
#define KTOP   256

typedef __attribute__((ext_vector_type(16))) __bf16         v16bf;
typedef __attribute__((ext_vector_type(8)))  float          v8f;
typedef __attribute__((ext_vector_type(8)))  unsigned short v8us;
typedef __attribute__((ext_vector_type(4)))  unsigned short v4us;

union FragA { v16bf bv; unsigned short us[16]; };

__device__ __forceinline__ unsigned short f32_to_bf16(float f) {
    unsigned u = __float_as_uint(f);
    unsigned rnd = 0x7FFFu + ((u >> 16) & 1u);   // round-to-nearest-even
    return (unsigned short)((u + rnd) >> 16);
}

// ---------------------------------------------------------------------------
// f32 -> bf16 (stored as u16) bulk convert, 4 elems/thread
// ---------------------------------------------------------------------------
__global__ void cvt_f32_bf16(const float* __restrict__ src,
                             unsigned short* __restrict__ dst, int n) {
    int i = (blockIdx.x * blockDim.x + threadIdx.x) * 4;
    if (i + 3 < n) {
        float4 v = *(const float4*)(src + i);
        v4us o;
        o[0] = f32_to_bf16(v.x); o[1] = f32_to_bf16(v.y);
        o[2] = f32_to_bf16(v.z); o[3] = f32_to_bf16(v.w);
        *(v4us*)(dst + i) = o;
    }
}

// ---------------------------------------------------------------------------
// Encode GEMM: C[n,f] = sum_d A[n,d]*B[f,d] + bias[f]   (NT GEMM, bf16 WMMA)
// block = 256 threads (8 waves). Wave tile: 16(M) x 128(N). Block: 128x128.
// ---------------------------------------------------------------------------
__global__ __launch_bounds__(256) void encode_gemm(
    const unsigned short* __restrict__ A,   // [N_ROWS, DIN] bf16
    const unsigned short* __restrict__ B,   // [DDICT,  DIN] bf16
    const float* __restrict__ bias,         // [DDICT]
    float* __restrict__ C)                  // [N_ROWS, DDICT] pre-act
{
    const int lane = threadIdx.x & 31;
    const int wave = threadIdx.x >> 5;
    const int rowBase = blockIdx.y * 128 + wave * 16;
    const int colBase = blockIdx.x * 128;

    const int m  = lane & 15;
    const int kh = lane >> 4;               // which K half this lane holds

    v8f acc[8];
#pragma unroll
    for (int t = 0; t < 8; ++t) acc[t] = {};

    const unsigned short* arow = A + (size_t)(rowBase + m) * DIN;

    for (int k0 = 0; k0 < DIN; k0 += 32) {
        FragA a;
        {   // A 16x32 as loaded here (contiguous K, same order as B): lane<16 -> K {0..15}; lane>=16 -> K {16..31}
            v8us c0 = *(const v8us*)(arow + k0 + kh * 16);
            v8us c1 = *(const v8us*)(arow + k0 + kh * 16 + 8);
#pragma unroll
            for (int i = 0; i < 8; ++i) { a.us[i] = c0[i]; a.us[i + 8] = c1[i]; }
        }
#pragma unroll
        for (int t = 0; t < 8; ++t) {
            // B 32x16: lane&15 = column, 16 contiguous K at offset (lane>>4)*16
            const unsigned short* brow =
                B + (size_t)(colBase + t * 16 + m) * DIN + k0 + kh * 16;
            v16bf b = *(const v16bf*)brow;
            acc[t] = __builtin_amdgcn_wmma_f32_16x16x32_bf16(
                false, a.bv, false, b, (short)0, acc[t], false, false);
        }
    }

    // C layout: VGPR i -> M = (lane>>4)*8 + i, N = lane&15
    const int col = lane & 15;
    const int mb  = (lane >> 4) * 8;
#pragma unroll
    for (int t = 0; t < 8; ++t) {
        const int c  = colBase + t * 16 + col;
        const float bv = bias[c];
#pragma unroll
        for (int i = 0; i < 8; ++i)
            C[(size_t)(rowBase + mb + i) * DDICT + c] = acc[t][i] + bv;
    }
}

// ---------------------------------------------------------------------------
// Per-row top-K threshold (radix select in LDS) + scatter-keep + ReLU, in place
// One 256-thread workgroup per row; 32 KB LDS keys.
// ---------------------------------------------------------------------------
__global__ __launch_bounds__(256) void topk_relu(float* __restrict__ H) {
    __shared__ unsigned keys[DDICT];
    __shared__ unsigned s_cnt;
    float* h = H + (size_t)blockIdx.x * DDICT;
    const int tid = threadIdx.x;

    for (int i = tid; i < DDICT; i += 256) {
        unsigned b = __float_as_uint(h[i]);
        keys[i] = (b & 0x80000000u) ? ~b : (b | 0x80000000u);  // monotonic key
    }
    __syncthreads();

    unsigned prefix = 0u, mask = 0u, kRemain = KTOP;
    for (int bit = 31; bit >= 0; --bit) {
        const unsigned tb = 1u << bit;
        const unsigned tp = prefix | tb;
        const unsigned tm = mask | tb;
        if (tid == 0) s_cnt = 0u;
        __syncthreads();
        unsigned local = 0;
        for (int i = tid; i < DDICT; i += 256)
            local += ((keys[i] & tm) == tp) ? 1u : 0u;
        atomicAdd(&s_cnt, local);                // LDS ds_add
        __syncthreads();
        unsigned c = s_cnt;
        if (c >= kRemain) prefix = tp; else kRemain -= c;
        mask = tm;
        __syncthreads();
    }
    // prefix == key of the K-th largest value; keep >= threshold, then ReLU
    for (int i = tid; i < DDICT; i += 256) {
        float v = h[i];
        h[i] = (keys[i] >= prefix) ? fmaxf(v, 0.f) : 0.f;
    }
}

// ---------------------------------------------------------------------------
// Decode GEMM: C[n,d] = sum_f A[n,f]*B[d,f] + bias[d]
// A (hidden) is f32, converted to bf16 on the fly per fragment.
// ---------------------------------------------------------------------------
__global__ __launch_bounds__(256) void decode_gemm(
    const float* __restrict__ A,            // [N_ROWS, DDICT] f32 hidden
    const unsigned short* __restrict__ B,   // [DIN, DDICT] bf16
    const float* __restrict__ bias,         // [DIN]
    float* __restrict__ C)                  // [N_ROWS, DIN]
{
    const int lane = threadIdx.x & 31;
    const int wave = threadIdx.x >> 5;
    const int rowBase = blockIdx.y * 128 + wave * 16;
    const int colBase = blockIdx.x * 128;

    const int m  = lane & 15;
    const int kh = lane >> 4;

    v8f acc[8];
#pragma unroll
    for (int t = 0; t < 8; ++t) acc[t] = {};

    const float* arow = A + (size_t)(rowBase + m) * DDICT;

    for (int k0 = 0; k0 < DDICT; k0 += 32) {
        FragA a;
        {
            v8f c0 = *(const v8f*)(arow + k0 + kh * 16);
            v8f c1 = *(const v8f*)(arow + k0 + kh * 16 + 8);
#pragma unroll
            for (int i = 0; i < 8; ++i) {
                a.us[i]     = f32_to_bf16(c0[i]);
                a.us[i + 8] = f32_to_bf16(c1[i]);
            }
        }
#pragma unroll
        for (int t = 0; t < 8; ++t) {
            const unsigned short* brow =
                B + (size_t)(colBase + t * 16 + m) * DDICT + k0 + kh * 16;
            v16bf b = *(const v16bf*)brow;
            acc[t] = __builtin_amdgcn_wmma_f32_16x16x32_bf16(
                false, a.bv, false, b, (short)0, acc[t], false, false);
        }
    }

    const int col = lane & 15;
    const int mb  = (lane >> 4) * 8;
#pragma unroll
    for (int t = 0; t < 8; ++t) {
        const int c  = colBase + t * 16 + col;
        const float bv = bias[c];
#pragma unroll
        for (int i = 0; i < 8; ++i)
            C[(size_t)(rowBase + mb + i) * DIN + c] = acc[t][i] + bv;
    }
}

// ---------------------------------------------------------------------------
extern "C" void kernel_launch(void* const* d_in, const int* in_sizes, int n_in,
                              void* d_out, int out_size, void* d_ws, size_t ws_size,
                              hipStream_t stream) {
    const float* x     = (const float*)d_in[0];   // [16384, 512]
    const float* W_enc = (const float*)d_in[1];   // [8192, 512]
    const float* b_enc = (const float*)d_in[2];   // [8192]
    const float* W_dec = (const float*)d_in[3];   // [512, 8192]
    const float* b_dec = (const float*)d_in[4];   // [512]

    float* recon  = (float*)d_out;                         // [16384, 512]
    float* hidden = recon + (size_t)N_ROWS * DIN;          // [16384, 8192]

    // bf16 scratch: x (16 MB) + W_enc (8 MB) + W_dec (8 MB) = 32 MB of d_ws
    unsigned short* x_bf    = (unsigned short*)d_ws;
    unsigned short* Wenc_bf = x_bf + (size_t)N_ROWS * DIN;
    unsigned short* Wdec_bf = Wenc_bf + (size_t)DDICT * DIN;

    {
        int n = N_ROWS * DIN;                       // 8,388,608
        cvt_f32_bf16<<<(n / 4 + 255) / 256, 256, 0, stream>>>(x, x_bf, n);
        n = DDICT * DIN;                            // 4,194,304
        cvt_f32_bf16<<<(n / 4 + 255) / 256, 256, 0, stream>>>(W_enc, Wenc_bf, n);
        cvt_f32_bf16<<<(n / 4 + 255) / 256, 256, 0, stream>>>(W_dec, Wdec_bf, n);
    }

    encode_gemm<<<dim3(DDICT / 128, N_ROWS / 128), 256, 0, stream>>>(
        x_bf, Wenc_bf, b_enc, hidden);

    topk_relu<<<N_ROWS, 256, 0, stream>>>(hidden);

    decode_gemm<<<dim3(DIN / 128, N_ROWS / 128), 256, 0, stream>>>(
        hidden, Wdec_bf, b_dec, recon);
}
